// TemporalViewModel_1090921693567
// MI455X (gfx1250) — matrix operand order repaired and checked by reading, not compiled, not run on hardware
//
#include <hip/hip_runtime.h>

// ---------------------------------------------------------------------------
// Fused GRU (T=24, N=65536 sequences, F=H=32) for gfx1250 / MI455X.
//
// Roofline: ~19.4 GFLOP vs ~201 MB of x traffic -> HBM-bound (~8.6 us floor
// at 23.3 TB/s). One kernel: x read once (coalesced b128 + global_prefetch),
// h resident in registers for all 24 steps, all matmuls via
// v_wmma_f32_16x16x32_f16 (K=32 == F == H; f32 accumulate).
//
// v3 changes (from round-2 asm): the `volatile` LDS fragment reads lowered
// to serialized flat_load_b128 scope:SYS (generic pointer + volatile) --
// terrible. Now: plain LDS reads (lower to ds_load_b128 with partial
// s_wait_dscnt interleaving) and an empty asm memory clobber at the top of
// the loop body to stop LICM from hoisting the 12 loop-invariant weight
// fragments (96 VGPRs) back into registers.
// ---------------------------------------------------------------------------

typedef __attribute__((ext_vector_type(16))) _Float16 v16h;
typedef __attribute__((ext_vector_type(8)))  _Float16 v8h;
typedef __attribute__((ext_vector_type(8)))  float    v8f;

static constexpr int kT    = 24;
static constexpr int kNPix = 16 * 64 * 64;   // B*X*Y = 65536
static constexpr int kS    = 16;             // output size

__device__ __forceinline__ float fast_sigmoid(float x) {
    return __builtin_amdgcn_rcpf(1.0f + __builtin_amdgcn_exp2f(-1.4426950408889634f * x));
}
__device__ __forceinline__ float fast_tanh(float x) {
    return 1.0f - 2.0f * __builtin_amdgcn_rcpf(1.0f + __builtin_amdgcn_exp2f(2.8853900817779268f * x));
}
__device__ __forceinline__ v8f splat8(float x) {
    v8f r;
    #pragma unroll
    for (int i = 0; i < 8; ++i) r[i] = x;
    return r;
}
__device__ __forceinline__ v8f wmma16(v16h a, v16h b, v8f c) {
    return __builtin_amdgcn_wmma_f32_16x16x32_f16(false, a, false, b, (short)0, c, false, false);
}
// plain LDS fragment read -> ds_load_b128 x2 (DScnt-tracked, batchable)
__device__ __forceinline__ v16h ldsfrag(const _Float16* p) {
    v8h a = *(const v8h*)p;
    v8h b = *(const v8h*)(p + 16);
    return __builtin_shufflevector(a, b, 0, 1, 2, 3, 4, 5, 6, 7, 8, 9, 10, 11, 12, 13, 14, 15);
}

__global__ __launch_bounds__(256)
void gru_fused_kernel(const float* __restrict__ spatial,
                      const float* __restrict__ met,
                      const float* __restrict__ ctx,
                      const float* __restrict__ Wih,   // (96,32) row-major
                      const float* __restrict__ Whh,   // (96,32) row-major
                      const float* __restrict__ bih,   // (96)
                      const float* __restrict__ bhh,   // (96)
                      const float* __restrict__ Wout,  // (32,16) row-major
                      float* __restrict__ out)         // (N,16)
{
    // f16 weight copies shared by all 8 waves (B-fragment source)
    __shared__ __align__(32) _Float16 wih_lds[96 * 32];   // 6 KB
    __shared__ __align__(32) _Float16 whh_lds[96 * 32];   // 6 KB
    // wave-private tile for the per-step h C-layout -> A-layout transpose
    __shared__ __align__(16) _Float16 hlds[8][16][32];    // 8 KB

    const int tid  = threadIdx.x;
    const int wave = tid >> 5;
    const int lane = tid & 31;
    const int mrow = lane & 15;   // M (pixel-in-tile) / N (col) index
    const int hi   = lane >> 4;   // lane half-select

    // ---- stage f16 weights to LDS (once per block) -----------------------
    for (int i = tid; i < 96 * 32; i += 256) {
        wih_lds[i] = (_Float16)Wih[i];
        whh_lds[i] = (_Float16)Whh[i];
    }
    __syncthreads();

    const int pixBase = (blockIdx.x * 8 + wave) * 16;
    const int p       = pixBase + mrow;      // this lane's pixel row

    // B layout (f16 32x16): lane: N = lane&15; j=0..7 -> K = hi*8+j, j=8..15 -> K = 16+hi*8+(j-8).
    // Fragment for gate-tile g = 2x16B at wX_lds + g*512 + mrow*32 + hi*8 (+16).
    const _Float16* wiB = wih_lds + mrow * 32 + hi * 8;
    const _Float16* whB = whh_lds + mrow * 32 + hi * 8;

    // biases (per-lane scalars)
    float biasRZ[4], biasIN[2], biasHN[2];
    #pragma unroll
    for (int g = 0; g < 6; ++g) {
        const int col = g * 16 + mrow;
        const float bi = bih[col], bh = bhh[col];
        if (g < 4)      biasRZ[g] = bi + bh;
        else { biasIN[g - 4] = bi; biasHN[g - 4] = bh; }
    }
    // output projection B fragment (kept in registers; only 8 VGPRs)
    v16h WoB;
    #pragma unroll
    for (int j = 0; j < 16; ++j)
        WoB[j] = (_Float16)Wout[(hi * 8 + j + (j & 8)) * 16 + mrow];

    // hidden state, f32, WMMA C/D layout: tile0 = units 0..15, tile1 = 16..31
    v8f hc0 = splat8(0.0f), hc1 = splat8(0.0f);
    v16h hA = {};   // h as f16 A fragment (zero at t=0 => gh = bias)

    // A layout (f16 16x32): lane: M = lane&15; j=0..7 -> K = hi*8+j,
    // j=8..15 -> K = 16+hi*8+(j-8). Feature map: K<16 spatial, [16,24) met,
    // [24,32) ctx => lanes 0-15: spatial[0..7]+met[0..7];
    //               lanes 16-31: spatial[8..15]+ctx[0..7].
    const float* xs = spatial + (size_t)p * 16 + hi * 8;
    const float* xm = (hi ? ctx : met) + (size_t)p * 8;
    const size_t strideS = (size_t)kNPix * 16;
    const size_t strideM = (size_t)kNPix * 8;

    for (int t = 0; t < kT; ++t) {
        // compiler-level memory barrier: emits nothing, but prevents LICM
        // from hoisting the 12 loop-invariant LDS weight fragments into
        // ~96 resident VGPRs (keeps them as per-step ds_load_b128).
        asm volatile("" ::: "memory");

        const float* s0 = xs + (size_t)t * strideS;
        const float* m0 = xm + (size_t)t * strideM;
        if (t + 1 < kT) {   // global_prefetch_b8 for next timestep's x
            __builtin_prefetch(s0 + strideS, 0, 1);
            __builtin_prefetch(m0 + strideM, 0, 1);
        }

        // x A fragment: two 32B contiguous runs per lane, cvt_pk to f16
        v16h xA;
        #pragma unroll
        for (int j = 0; j < 8; ++j) xA[j] = (_Float16)s0[j];
        #pragma unroll
        for (int j = 0; j < 8; ++j) xA[8 + j] = (_Float16)m0[j];

        // gh_n first (depends only on hA)
        v8f ghn0 = wmma16(hA, ldsfrag(whB + 4 * 512), splat8(biasHN[0]));
        v8f ghn1 = wmma16(hA, ldsfrag(whB + 5 * 512), splat8(biasHN[1]));
        // r, z: x-proj and h-proj chained into one accumulator
        v8f aR0 = wmma16(xA, ldsfrag(wiB + 0 * 512), splat8(biasRZ[0]));
        aR0     = wmma16(hA, ldsfrag(whB + 0 * 512), aR0);
        v8f aR1 = wmma16(xA, ldsfrag(wiB + 1 * 512), splat8(biasRZ[1]));
        aR1     = wmma16(hA, ldsfrag(whB + 1 * 512), aR1);
        v8f aZ0 = wmma16(xA, ldsfrag(wiB + 2 * 512), splat8(biasRZ[2]));
        aZ0     = wmma16(hA, ldsfrag(whB + 2 * 512), aZ0);
        v8f aZ1 = wmma16(xA, ldsfrag(wiB + 3 * 512), splat8(biasRZ[3]));
        aZ1     = wmma16(hA, ldsfrag(whB + 3 * 512), aZ1);

        // n-gate: C operand = b_ih_n + r * gh_n, then one WMMA adds x*W_ih_n
        v8f cn0, cn1;
        #pragma unroll
        for (int v = 0; v < 8; ++v) {
            cn0[v] = biasIN[0] + fast_sigmoid(aR0[v]) * ghn0[v];
            cn1[v] = biasIN[1] + fast_sigmoid(aR1[v]) * ghn1[v];
        }
        v8f n0 = wmma16(xA, ldsfrag(wiB + 4 * 512), cn0);
        v8f n1 = wmma16(xA, ldsfrag(wiB + 5 * 512), cn1);

        // z gate + state update
        #pragma unroll
        for (int v = 0; v < 8; ++v) {
            float z0 = fast_sigmoid(aZ0[v]);
            float t0 = fast_tanh(n0[v]);
            hc0[v]   = t0 + z0 * (hc0[v] - t0);

            float z1 = fast_sigmoid(aZ1[v]);
            float t1 = fast_tanh(n1[v]);
            hc1[v]   = t1 + z1 * (hc1[v] - t1);
        }

        // C-layout -> A-layout transpose through wave-private LDS.
        // LDS ops are in-order within a wave (DScnt); no barrier needed.
        #pragma unroll
        for (int v = 0; v < 8; ++v) {
            hlds[wave][v + 8 * hi][mrow]      = (_Float16)hc0[v];
            hlds[wave][v + 8 * hi][16 + mrow] = (_Float16)hc1[v];
        }
        const v8h* hrow = (const v8h*)&hlds[wave][mrow][0];  // 4x 16B chunks
        v8h c1 = hrow[hi];        // K = hi*8 .. hi*8+7
        v8h c2 = hrow[2 + hi];    // K = 16+hi*8 .. +7
        #pragma unroll
        for (int j = 0; j < 8; ++j) { hA[j] = c1[j]; hA[8 + j] = c2[j]; }
    }

    // ---- output projection: out(16x16) = h(16x32) x Wout(32x16) ----------
    v8f o = wmma16(hA, WoB, splat8(0.0f));
    #pragma unroll
    for (int v = 0; v < 8; ++v)
        out[(size_t)(pixBase + v + 8 * hi) * kS + mrow] = o[v];
}

extern "C" void kernel_launch(void* const* d_in, const int* in_sizes, int n_in,
                              void* d_out, int out_size, void* d_ws, size_t ws_size,
                              hipStream_t stream) {
    (void)in_sizes; (void)n_in; (void)out_size; (void)d_ws; (void)ws_size;
    const float* spatial = (const float*)d_in[0];
    const float* met     = (const float*)d_in[1];
    const float* ctx     = (const float*)d_in[2];
    const float* Wih     = (const float*)d_in[3];
    const float* Whh     = (const float*)d_in[4];
    const float* bih     = (const float*)d_in[5];
    const float* bhh     = (const float*)d_in[6];
    const float* Wout    = (const float*)d_in[7];
    float* out           = (float*)d_out;

    // 16 pixels/wave * 8 waves/block = 128 pixels/block -> 512 blocks
    dim3 grid(kNPix / 128), block(256);
    gru_fused_kernel<<<grid, block, 0, stream>>>(spatial, met, ctx, Wih, Whh,
                                                 bih, bhh, Wout, out);
}
